// MultiHeadRelativeAttention_69801808495017
// MI455X (gfx1250) — matrix operand order repaired and checked by reading, not compiled, not run on hardware
//
#include <hip/hip_runtime.h>
#include <hip/hip_bf16.h>

// ---------------------------------------------------------------------------
// MultiHeadRelativeAttention for MI455X (gfx1250, wave32, WMMA bf16,
// async global->LDS staging). B=4, T=2048, H=8, D=32, HEAD_SIZE=256.
// ---------------------------------------------------------------------------

#define Bsz  4
#define Tlen 2048
#define NH   8
#define HD   32
#define NE   256
#define HS   256
#define N3   768

typedef __attribute__((ext_vector_type(16))) __bf16 v16bf;
typedef __attribute__((ext_vector_type(8)))  __bf16 v8bf;
typedef __attribute__((ext_vector_type(8)))  float  v8f;

union ABf { v16bf v; v8bf h[2]; __bf16 e[16]; };
static_assert(sizeof(ABf) == 32, "fragment must be 8 VGPRs");

__device__ __forceinline__ v8f wmma_bf16(v16bf a, v16bf b, v8f c) {
    // v_wmma_f32_16x16x32_bf16  D = A(16x32) * B(32x16) + C(16x16 f32)
    return __builtin_amdgcn_wmma_f32_16x16x32_bf16(false, a, false, b,
                                                   (short)0, c, false, false);
}

__device__ __forceinline__ void cvt4(__bf16* d, float4 f) {
    d[0] = (__bf16)f.x; d[1] = (__bf16)f.y; d[2] = (__bf16)f.z; d[3] = (__bf16)f.w;
}

// LDS byte offset of a generic pointer into __shared__ (ISA 10.2: addr[31:0])
__device__ __forceinline__ unsigned lds_off_of(const void* p) {
    return (unsigned)(unsigned long long)p;
}

// One b128 async DMA lane-op: LDS[lds_off] <- global[gptr]   (ASYNCcnt)
__device__ __forceinline__ void async_b128(unsigned lds_off, const void* gptr) {
    asm volatile("global_load_async_to_lds_b128 %0, %1, off"
                 :: "v"(lds_off), "v"(gptr) : "memory");
}

// Stage one 32x32 bf16 K tile (2KB contiguous) + one 32x32 bf16 V tile
// (32 rows x 64B, row stride Tlen*2 bytes) into a per-wave LDS buffer.
// 8 async b128 instructions per wave.
__device__ __forceinline__ void stage_kv(const __bf16* Kg, const __bf16* Vg,
                                         const __bf16* buf, int lane)
{
    const unsigned kl = lds_off_of(buf) + lane * 16;
    const char* kgb = (const char*)Kg + lane * 16;
#pragma unroll
    for (int j = 0; j < 4; ++j)
        async_b128(kl + j * 512, kgb + j * 512);

    const unsigned vl = lds_off_of(buf + 1024);
#pragma unroll
    for (int j = 0; j < 4; ++j) {
        const int cc  = j * 32 + lane;      // 16B chunk id, 128 chunks
        const int row = cc >> 2, col = cc & 3;
        async_b128(vl + cc * 16,
                   (const char*)Vg + (size_t)row * (Tlen * 2) + col * 16);
    }
}

// ---------------------------------------------------------------------------
// K0: repack weights / pe to bf16, K-contiguous for B-fragment b128 loads
// ---------------------------------------------------------------------------
__global__ void prep_kernel(const float* __restrict__ w_attn,
                            const float* __restrict__ pe,
                            const float* __restrict__ w_proj,
                            __bf16* __restrict__ wT,
                            __bf16* __restrict__ peb,
                            __bf16* __restrict__ wpT)
{
    const int i = blockIdx.x * 256 + threadIdx.x;
    if (i < N3 * NE) {                       // wT[c][e] = w_attn[e][c]
        const int c = i >> 8, e = i & 255;
        wT[(size_t)c * NE + e] = (__bf16)w_attn[(size_t)e * N3 + c];
    }
    if (i < HS * NE) {                       // wpT[c][k] = w_proj[k][c]
        const int c = i >> 8, k = i & 255;
        wpT[(size_t)c * HS + k] = (__bf16)w_proj[(size_t)k * NE + c];
    }
    if (i < NH * Tlen * HD)                  // peb = bf16(pos_embs)
        peb[i] = (__bf16)pe[i];
}

// ---------------------------------------------------------------------------
// K1: qkv = x @ w_attn  (8192 x 768 x 256).
// Block = 8 waves sharing one 16-row M panel of x staged once to LDS (bf16).
// Each wave owns one 16-col tile; outputs scattered to Qb/Kb (t-major) and
// Vt (d-major) so attention fragments load as b128.
// ---------------------------------------------------------------------------
__global__ __launch_bounds__(256) void qkv_gemm(const float*  __restrict__ x,
                                                const __bf16* __restrict__ wT,
                                                __bf16* __restrict__ Qb,
                                                __bf16* __restrict__ Kb,
                                                __bf16* __restrict__ Vt)
{
    __shared__ __align__(16) __bf16 sA[16 * NE];           // 8 KB

    const int tid  = threadIdx.x;
    const int lane = tid & 31,  wave = tid >> 5;
    const int ln = lane & 15,   lh = lane >> 4;
    const int mt = blockIdx.x / 6, ntg = blockIdx.x % 6;   // 512 x 6 blocks
    const int m0 = mt << 4;
    const int c0 = (ntg * 8 + wave) << 4;

    {   // cooperative A-panel stage: 16x256 f32 -> bf16, 16 elems/thread
        const int row = tid >> 4, ch = (tid & 15) << 4;
        const float* p = x + (size_t)(m0 + row) * NE + ch;
        __bf16* d = sA + row * NE + ch;
        cvt4(d + 0,  *(const float4*)(p));
        cvt4(d + 4,  *(const float4*)(p + 4));
        cvt4(d + 8,  *(const float4*)(p + 8));
        cvt4(d + 12, *(const float4*)(p + 12));
    }
    __syncthreads();

    v8f acc = {0.f,0.f,0.f,0.f,0.f,0.f,0.f,0.f};
    const __bf16* ar = sA + ln * NE + lh * 16;
    const __bf16* wr = wT + (size_t)(c0 + ln) * NE + lh * 16;

#pragma unroll
    for (int k = 0; k < NE; k += 32) {
        ABf a, bfr;
        a.h[0]   = *(const v8bf*)(ar + k);
        a.h[1]   = *(const v8bf*)(ar + k + 8);
        bfr.h[0] = *(const v8bf*)(wr + k);
        bfr.h[1] = *(const v8bf*)(wr + k + 8);
        acc = wmma_bf16(a.v, bfr.v, acc);
    }

    const int g  = c0 >> 8;                      // 0=K, 1=Q, 2=V
    const int hh = (c0 & 255) >> 5;              // head
    const int d  = (c0 & 31) + ln;               // dim within head
#pragma unroll
    for (int r = 0; r < 8; ++r) {
        const int mrow = m0 + r + lh * 8;
        const int bb = mrow >> 11, t = mrow & (Tlen - 1);
        const int bhh = bb * NH + hh;
        const __bf16 v = (__bf16)acc[r];
        if (g == 0)      Kb[((size_t)bhh * Tlen + t) * HD + d] = v;
        else if (g == 1) Qb[((size_t)bhh * Tlen + t) * HD + d] = v;
        else             Vt[((size_t)bhh * HD + d) * Tlen + t] = v;
    }
}

// ---------------------------------------------------------------------------
// K2: fused flash attention with relative-position bias.
// 128 threads = 4 waves; one wave owns a 16-row t-tile and loops s in
// 32-blocks up to the diagonal. K/V tiles are streamed with double-buffered
// global_load_async_to_lds_b128 (ASYNCcnt) while WMMAs run.
// Per block: 2 QK wmma + 3 Qpe wmma (skew band) + online softmax + 2 PV wmma.
// ---------------------------------------------------------------------------
#define WV 4
__global__ __launch_bounds__(128) void attn_kernel(const __bf16* __restrict__ Qb,
                                                   const __bf16* __restrict__ Kb,
                                                   const __bf16* __restrict__ Vt,
                                                   const __bf16* __restrict__ peb,
                                                   __bf16* __restrict__ attn_b)
{
    __shared__ __align__(16) float  s_rel[WV][16 * 48];   // 12 KB skew scratch
    __shared__ __align__(16) __bf16 s_p[WV][16 * 32];     //  4 KB P staging
    __shared__ __align__(16) __bf16 s_kv[WV][2][2048];    // 32 KB K|V dbl-buf

    const int lane = threadIdx.x & 31;
    const int wave = threadIdx.x >> 5;
    const int ln = lane & 15, lh = lane >> 4;
    const int task = blockIdx.x * WV + wave;            // 0..4095
    const int bh = task >> 7;                           // (b,h)
    const int t0 = (task & 127) << 4;
    const int b  = bh >> 3, h = bh & 7;
    float*  relw = s_rel[wave];
    __bf16* pw   = s_p[wave];

    const __bf16* Kb_bh = Kb  + (size_t)bh * Tlen * HD;
    const __bf16* Vt_bh = Vt  + (size_t)bh * HD * Tlen;
    const __bf16* pe_h  = peb + (size_t)h  * Tlen * HD;

    // Q A-fragment (rows t0..t0+15, K = head dim 0..31)
    ABf qf;
    {
        const __bf16* qr = Qb + ((size_t)bh * Tlen + t0 + ln) * HD + lh * 16;
        qf.h[0] = *(const v8bf*)qr;
        qf.h[1] = *(const v8bf*)(qr + 8);
    }

    float m[8], l[8];
    v8f O0 = {0.f,0.f,0.f,0.f,0.f,0.f,0.f,0.f}, O1 = O0;
    const v8f vz = O0;
#pragma unroll
    for (int r = 0; r < 8; ++r) { m[r] = -1e30f; l[r] = 0.f; }

    const int nblk = (t0 + 47) >> 5;                    // causal: s <= t0+15

    // preload block 0 K/V tiles
    stage_kv(Kb_bh, Vt_bh, s_kv[wave][0], lane);

    for (int blk = 0; blk < nblk; ++blk) {
        const int sb = blk << 5;
        const int u0 = Tlen - 16 - t0 + sb;

        if (blk + 1 < nblk) {
            // kick off next block's DMA, then wait only for current block
            stage_kv(Kb_bh + (size_t)(sb + 32) * HD, Vt_bh + (sb + 32),
                     s_kv[wave][(blk + 1) & 1], lane);
            {   // prefetch next pe band into caches (global_prefetch_b8)
                int un = u0 + 32 + lane;
                if (un > Tlen - 1) un = Tlen - 1;
                __builtin_prefetch(pe_h + (size_t)un * HD, 0, 1);
            }
            asm volatile("s_wait_asynccnt 0x8" ::: "memory");
        } else {
            asm volatile("s_wait_asynccnt 0x0" ::: "memory");
        }

        // fragments from staged LDS tiles (ds_load_b128)
        const __bf16* kt = s_kv[wave][blk & 1];
        const __bf16* vt = kt + 1024;
        ABf kf0, kf1, vf0, vf1;
        {
            const __bf16* kr = kt + ln * 32 + lh * 16;
            kf0.h[0] = *(const v8bf*)kr;      kf0.h[1] = *(const v8bf*)(kr + 8);
            kr += 16 * 32;
            kf1.h[0] = *(const v8bf*)kr;      kf1.h[1] = *(const v8bf*)(kr + 8);
            const __bf16* vr = vt + ln * 32 + lh * 16;
            vf0.h[0] = *(const v8bf*)vr;      vf0.h[1] = *(const v8bf*)(vr + 8);
            vr += 16 * 32;
            vf1.h[0] = *(const v8bf*)vr;      vf1.h[1] = *(const v8bf*)(vr + 8);
        }

        v8f S0 = wmma_bf16(qf.v, kf0.v, vz);
        v8f S1 = wmma_bf16(qf.v, kf1.v, vz);

        // Relative bias: Ppe[t,u] = Q[t].pe[u], u in [u0,u0+47] (skew band)
#pragma unroll
        for (int j = 0; j < 3; ++j) {
            int u = u0 + j * 16 + ln;
            if (u > Tlen - 1) u = Tlen - 1;             // masked region only
            const __bf16* pr = pe_h + (size_t)u * HD + lh * 16;
            ABf pf;
            pf.h[0] = *(const v8bf*)pr; pf.h[1] = *(const v8bf*)(pr + 8);
            v8f P = wmma_bf16(qf.v, pf.v, vz);
#pragma unroll
            for (int r = 0; r < 8; ++r)
                relw[(r + lh * 8) * 48 + j * 16 + ln] = P[r];
        }

        // Skew extraction + scale + causal mask. n = 15-row+ds in [0,46].
        float sc0[8], sc1[8];
#pragma unroll
        for (int r = 0; r < 8; ++r) {
            const int row = r + lh * 8;
            const int t = t0 + row;
            const float r0 = relw[row * 48 + 15 - row + ln];
            const float r1 = relw[row * 48 + 31 - row + ln];
            float a0 = (S0[r] + r0) * 0.17677669529663688f;   // 1/sqrt(32)
            float a1 = (S1[r] + r1) * 0.17677669529663688f;
            if (sb + ln      > t) a0 += -1.0e9f;
            if (sb + 16 + ln > t) a1 += -1.0e9f;
            sc0[r] = a0; sc1[r] = a1;
        }

        // Online softmax (row = 16 lanes within half-wave; xor-butterflies)
#pragma unroll
        for (int r = 0; r < 8; ++r) {
            float xm = fmaxf(sc0[r], sc1[r]);
            xm = fmaxf(xm, __shfl_xor(xm, 1, 32));
            xm = fmaxf(xm, __shfl_xor(xm, 2, 32));
            xm = fmaxf(xm, __shfl_xor(xm, 4, 32));
            xm = fmaxf(xm, __shfl_xor(xm, 8, 32));
            const float mn = fmaxf(m[r], xm);
            const float sc = __expf(m[r] - mn);
            m[r] = mn;
            const float p0 = __expf(sc0[r] - mn);
            const float p1 = __expf(sc1[r] - mn);
            float ps = p0 + p1;
            ps += __shfl_xor(ps, 1, 32);
            ps += __shfl_xor(ps, 2, 32);
            ps += __shfl_xor(ps, 4, 32);
            ps += __shfl_xor(ps, 8, 32);
            l[r] = l[r] * sc + ps;
            O0[r] *= sc; O1[r] *= sc;
            const int row = r + lh * 8;
            pw[row * 32 + ln]      = (__bf16)p0;
            pw[row * 32 + 16 + ln] = (__bf16)p1;
        }

        // Reload P in A-fragment layout (same-wave LDS is in-order)
        ABf pfr;
        {
            const __bf16* pr = pw + ln * 32 + lh * 16;
            pfr.h[0] = *(const v8bf*)pr; pfr.h[1] = *(const v8bf*)(pr + 8);
        }
        O0 = wmma_bf16(pfr.v, vf0.v, O0);
        O1 = wmma_bf16(pfr.v, vf1.v, O1);
    }

    // Normalize and emit bf16 attention output [B, T, H*D]
#pragma unroll
    for (int r = 0; r < 8; ++r) {
        const float inv = 1.0f / l[r];
        const int t = t0 + r + lh * 8;
        __bf16* orow = attn_b + ((size_t)b * Tlen + t) * HS + h * HD;
        orow[ln]      = (__bf16)(O0[r] * inv);
        orow[16 + ln] = (__bf16)(O1[r] * inv);
    }
}

// ---------------------------------------------------------------------------
// K3: out = attn @ w_proj + b_proj  (8192 x 256 x 256), f32 output.
// Block = 8 waves sharing one staged 16x256 bf16 A panel.
// ---------------------------------------------------------------------------
__global__ __launch_bounds__(256) void proj_gemm(const __bf16* __restrict__ attn_b,
                                                 const __bf16* __restrict__ wpT,
                                                 const float*  __restrict__ b_proj,
                                                 float* __restrict__ out)
{
    __shared__ __align__(16) __bf16 sA[16 * HS];           // 8 KB

    const int tid  = threadIdx.x;
    const int lane = tid & 31,  wave = tid >> 5;
    const int ln = lane & 15,   lh = lane >> 4;
    const int mt = blockIdx.x >> 1, ntg = blockIdx.x & 1;  // 512 x 2 blocks
    const int m0 = mt << 4;
    const int c0 = (ntg * 8 + wave) << 4;

    {   // cooperative A-panel stage (already bf16): 32B/thread
        const int row = tid >> 4, ch = (tid & 15) << 4;
        const __bf16* src = attn_b + (size_t)(m0 + row) * HS + ch;
        *(v8bf*)(sA + row * HS + ch)     = *(const v8bf*)src;
        *(v8bf*)(sA + row * HS + ch + 8) = *(const v8bf*)(src + 8);
    }
    __syncthreads();

    v8f acc = {0.f,0.f,0.f,0.f,0.f,0.f,0.f,0.f};
    const __bf16* ar = sA + ln * HS + lh * 16;
    const __bf16* wr = wpT + (size_t)(c0 + ln) * HS + lh * 16;

#pragma unroll
    for (int k = 0; k < HS; k += 32) {
        ABf a, bfr;
        a.h[0]   = *(const v8bf*)(ar + k);
        a.h[1]   = *(const v8bf*)(ar + k + 8);
        bfr.h[0] = *(const v8bf*)(wr + k);
        bfr.h[1] = *(const v8bf*)(wr + k + 8);
        acc = wmma_bf16(a.v, bfr.v, acc);
    }

    const float bias = b_proj[c0 + ln];
#pragma unroll
    for (int r = 0; r < 8; ++r)
        out[(size_t)(m0 + r + lh * 8) * NE + c0 + ln] = acc[r] + bias;
}

// ---------------------------------------------------------------------------
extern "C" void kernel_launch(void* const* d_in, const int* in_sizes, int n_in,
                              void* d_out, int out_size, void* d_ws, size_t ws_size,
                              hipStream_t stream)
{
    const float* x      = (const float*)d_in[0];
    const float* w_attn = (const float*)d_in[1];
    const float* pos    = (const float*)d_in[2];
    const float* w_proj = (const float*)d_in[3];
    const float* b_proj = (const float*)d_in[4];
    float* out = (float*)d_out;

    char* ws = (char*)d_ws;
    size_t off = 0;
    auto take = [&](size_t bytes) -> char* {
        char* p = ws + off;
        off += (bytes + 255) & ~(size_t)255;
        return p;
    };
    __bf16* Qb     = (__bf16*)take((size_t)Bsz * NH * Tlen * HD * 2);  // 4 MB
    __bf16* Kb     = (__bf16*)take((size_t)Bsz * NH * Tlen * HD * 2);  // 4 MB
    __bf16* Vt     = (__bf16*)take((size_t)Bsz * NH * Tlen * HD * 2);  // 4 MB
    __bf16* peb    = (__bf16*)take((size_t)NH * Tlen * HD * 2);        // 1 MB
    __bf16* wT     = (__bf16*)take((size_t)N3 * NE * 2);
    __bf16* wpT    = (__bf16*)take((size_t)HS * NE * 2);
    __bf16* attn_b = (__bf16*)take((size_t)Bsz * Tlen * HS * 2);       // 4 MB

    prep_kernel<<<2048, 256, 0, stream>>>(w_attn, pos, w_proj, wT, peb, wpT);
    qkv_gemm  <<<3072, 256, 0, stream>>>(x, wT, Qb, Kb, Vt);
    attn_kernel<<<1024, 128, 0, stream>>>(Qb, Kb, Vt, peb, attn_b);
    proj_gemm <<<1024, 256, 0, stream>>>(attn_b, wpT, b_proj, out);
}
